// Attn_Module_27900107554849
// MI455X (gfx1250) — matrix operand order repaired and checked by reading, not compiled, not run on hardware
//
#include <hip/hip_runtime.h>
#include <hip/hip_bf16.h>

typedef __bf16 bf16;
typedef __attribute__((ext_vector_type(4)))  __bf16 v4bf;
typedef __attribute__((ext_vector_type(8)))  __bf16 v8bf;
typedef __attribute__((ext_vector_type(16))) __bf16 v16bf;
typedef __attribute__((ext_vector_type(8)))  float  v8f;
typedef __attribute__((ext_vector_type(4)))  unsigned int u32x4;
typedef __attribute__((ext_vector_type(8)))  int i32x8;
typedef __attribute__((ext_vector_type(4)))  int i32x4;

#define B_   4
#define C_   512
#define C8_  64
#define N_   4096

#if __has_builtin(__builtin_amdgcn_tensor_load_to_lds) && \
    __has_builtin(__builtin_amdgcn_s_wait_tensorcnt)
#define USE_TDM 1
#endif

// ---------------------------------------------------------------------------
// WMMA fragment loaders (CDNA5 ISA 7.12.2, wave32), fragment-major LDS tiles.
// A tile stored [m][k] (ld mult of 8): lane (m=L%16, h=L/16) reads two aligned
//   16B chunks at k0+8h and k0+16+8h  -> 2x ds_load_b128.
// B tile stored TRANSPOSED [n][k] (ld mult of 8): lane (n=L%16, h=L/16) reads
//   two aligned 16B chunks at k0+8h and k0+16+8h -> 2x ds_load_b128.
// C/D 16x16 f32: lane -> N = L%16; VGPR r -> M = r + 8*(L/16)
// ---------------------------------------------------------------------------

__device__ __forceinline__ v16bf frag_A(const bf16* lds, int m0, int k0, int ld) {
    int lane = threadIdx.x & 31;
    int m = lane & 15, h = lane >> 4;
    const bf16* base = lds + (m0 + m) * ld + k0 + 8 * h;
    union { v16bf v; v8bf hlf[2]; } u;
    u.hlf[0] = *(const v8bf*)(base);
    u.hlf[1] = *(const v8bf*)(base + 16);
    return u.v;
}

__device__ __forceinline__ v16bf frag_B(const bf16* ldsT, int n0, int k0, int ld) {
    int lane = threadIdx.x & 31;
    int n = lane & 15, h = lane >> 4;
    const bf16* base = ldsT + (n0 + n) * ld + k0 + 8 * h;
    union { v16bf v; v8bf hlf[2]; } u;
    u.hlf[0] = *(const v8bf*)(base);
    u.hlf[1] = *(const v8bf*)(base + 16);
    return u.v;
}

__device__ __forceinline__ v8f wmma_bf16(v16bf a, v16bf b, v8f c) {
    return __builtin_amdgcn_wmma_f32_16x16x32_bf16(
        false, a, false, b, (short)0, c, false, false);
}

__device__ __forceinline__ v4bf pack4(float a, float b, float c, float d) {
    v4bf p; p[0] = (bf16)a; p[1] = (bf16)b; p[2] = (bf16)c; p[3] = (bf16)d;
    return p;
}

__device__ __forceinline__ unsigned lds_offset(const void* p) {
    // LDS flat addresses: low 32 bits are the LDS byte address (ISA 10.2)
    return (unsigned)(size_t)p;
}

// ---------------------------------------------------------------------------
// Kernel 1: f/g/h projections. Out[sel][b][c8][n](bf16) = W (64x512) @ X_b (512x4096)
// Grid (N/64, 3, B), 256 threads; 2 output tiles per wave; K loop of 32.
// ---------------------------------------------------------------------------
__global__ __launch_bounds__(256)
void proj_kernel(const float* __restrict__ x,
                 const float* __restrict__ Wf,
                 const float* __restrict__ Wg,
                 const float* __restrict__ Wh,
                 bf16* __restrict__ FGH) {
    const int jb  = blockIdx.x * 64;
    const int sel = blockIdx.y;
    const int b   = blockIdx.z;
    const float* W = (sel == 0) ? Wf : (sel == 1) ? Wg : Wh;
    const float* X = x + (size_t)b * C_ * N_;
    bf16* Out = FGH + ((size_t)sel * B_ + b) * C8_ * N_;

    __shared__ __align__(16) bf16 At [64 * 32];   // [c8][k]  ld 32
    __shared__ __align__(16) bf16 BtT[64 * 40];   // [j][k]   ld 40 (padded)
    __shared__ __align__(16) bf16 Obuf[64 * 64];  // [c8][j]  ld 64

    const int tid  = threadIdx.x;
    const int wave = tid >> 5;
    const int lane = tid & 31;
    const int col  = lane & 15, hh = lane >> 4;
    const int ct = wave >> 2;                     // c-tiles ct, ct+2
    const int jt = wave & 3;

    v8f acc0 = {0.f,0.f,0.f,0.f,0.f,0.f,0.f,0.f};
    v8f acc1 = acc0;

    for (int k0 = 0; k0 < C_; k0 += 32) {
        __syncthreads();
#pragma unroll
        for (int e = 0; e < 2; ++e) {             // A: 64x32 fp32 -> bf16
            int ch = tid + e * 256;
            int r = ch >> 3, c4 = (ch & 7) * 4;
            float4 w4 = *(const float4*)&W[(size_t)r * C_ + k0 + c4];
            *(v4bf*)&At[r * 32 + c4] = pack4(w4.x, w4.y, w4.z, w4.w);
        }
#pragma unroll
        for (int e = 0; e < 2; ++e) {             // B: 32x64, transposed scatter
            int ch = tid + e * 256;
            int r = ch >> 4, j4 = (ch & 15) * 4;
            float4 v = *(const float4*)&X[(size_t)(k0 + r) * N_ + jb + j4];
            BtT[(j4 + 0) * 40 + r] = (bf16)v.x;
            BtT[(j4 + 1) * 40 + r] = (bf16)v.y;
            BtT[(j4 + 2) * 40 + r] = (bf16)v.z;
            BtT[(j4 + 3) * 40 + r] = (bf16)v.w;
        }
        __syncthreads();
        v16bf bfrag = frag_B(BtT, jt * 16, 0, 40);
        acc0 = wmma_bf16(frag_A(At, ct * 16, 0, 32), bfrag, acc0);
        acc1 = wmma_bf16(frag_A(At, (ct + 2) * 16, 0, 32), bfrag, acc1);
    }

    __syncthreads();
#pragma unroll
    for (int r = 0; r < 8; ++r) {
        int m = r + 8 * hh;
        Obuf[(ct * 16 + m) * 64 + jt * 16 + col]       = (bf16)acc0[r];
        Obuf[((ct + 2) * 16 + m) * 64 + jt * 16 + col] = (bf16)acc1[r];
    }
    __syncthreads();
    {
        int c  = tid >> 2;
        int j0 = (tid & 3) * 16;
        const v8bf* src = (const v8bf*)&Obuf[c * 64 + j0];
        v8bf* dst = (v8bf*)&Out[(size_t)c * N_ + jb + j0];
        dst[0] = src[0];
        dst[1] = src[1];
    }
}

// ---------------------------------------------------------------------------
// Kernel 2: flash attention. Per (b, 64-query tile), loop 64 key tiles of 64:
//   S = F^T G (4 WMMA/wave) -> parallel online softmax -> O += H P (4 WMMA/wave).
// V tile loaded via Tensor Data Mover when available. Grid (N/64, B), 256 thr.
// ---------------------------------------------------------------------------
__global__ __launch_bounds__(256)
void attn_kernel(const bf16* __restrict__ FGH, float* __restrict__ SA8) {
    const int jb = blockIdx.x * 64;
    const int b  = blockIdx.y;
    const bf16* F = FGH + ((size_t)0 * B_ + b) * C8_ * N_;
    const bf16* G = FGH + ((size_t)1 * B_ + b) * C8_ * N_;
    const bf16* H = FGH + ((size_t)2 * B_ + b) * C8_ * N_;

    __shared__ __align__(16) bf16  GtT[64 * 72];  // [j][c]  ld 72 (padded)
    __shared__ __align__(16) bf16  FtT[64 * 72];  // [i][c]  ld 72
    __shared__ __align__(16) bf16  Ht [64 * 64];  // [c][i]  ld 64 (TDM layout)
    __shared__ __align__(16) bf16  PtT[64 * 72];  // [j][i]  ld 72
    __shared__ float Sm[64 * 65];                 // [j][i]  ld 65 (conflict-free)
    __shared__ float red[4][64];
    __shared__ float mrow[64], lrow[64], arow[64];

    const int tid  = threadIdx.x;
    const int wave = tid >> 5;
    const int lane = tid & 31;
    const int col  = lane & 15, hh = lane >> 4;
    const int srow = wave >> 2, scol = wave & 3;  // S tiles (srow, srow+2) x scol
    const int ct   = wave >> 2, jt   = wave & 3;  // O tiles rows ct, ct+2
    const int j    = tid & 63,  q    = tid >> 6;  // softmax: 4 threads/column

    // Q tile: resident for the whole kernel, stored transposed [j][c]
#pragma unroll
    for (int e = 0; e < 2; ++e) {
        int ch = tid + e * 256;
        int c = ch >> 3, j8 = (ch & 7) * 8;
        v8bf v = *(const v8bf*)&G[(size_t)c * N_ + jb + j8];
#pragma unroll
        for (int u = 0; u < 8; ++u) GtT[(j8 + u) * 72 + c] = v[u];
    }
    if (tid < 64) { mrow[tid] = -3.0e38f; lrow[tid] = 0.f; }

    v8f o0 = {0.f,0.f,0.f,0.f,0.f,0.f,0.f,0.f};
    v8f o1 = o0;

    for (int t = 0; t < N_; t += 64) {
        __syncthreads();
        // K tile -> FtT [i][c] (transposed scatter)
        {
            int c = tid >> 2, i16 = (tid & 3) * 16;
            v8bf va = *(const v8bf*)&F[(size_t)c * N_ + t + i16];
            v8bf vb = *(const v8bf*)&F[(size_t)c * N_ + t + i16 + 8];
#pragma unroll
            for (int u = 0; u < 8; ++u) FtT[(i16 + u) * 72 + c]     = va[u];
#pragma unroll
            for (int u = 0; u < 8; ++u) FtT[(i16 + 8 + u) * 72 + c] = vb[u];
        }
        // V tile -> Ht [c][i]: TDM 2-D tile load (64 rows x 64 elems, stride N)
#ifdef USE_TDM
        if (tid < 32) {
            unsigned long long ga = (unsigned long long)(size_t)(H + t);
            u32x4 g0; i32x8 g1;
            i32x4 g2 = {0, 0, 0, 0}, g3 = {0, 0, 0, 0};
            i32x8 g4 = {0, 0, 0, 0, 0, 0, 0, 0};
            g0[0] = 1u;                               // count=1, normal D#
            g0[1] = lds_offset(Ht);                   // lds_addr
            g0[2] = (unsigned)ga;                     // global_addr[31:0]
            g0[3] = (unsigned)((ga >> 32) & 0x01FFFFFFu) | 0x80000000u; // type=2
            g1[0] = 1 << 16;                          // data_size=2B
            g1[1] = (int)(4096u << 16);               // tensor_dim0 = N
            g1[2] = 64 << 16;                         // tensor_dim1 = 64 rows
            g1[3] = 64 << 16;                         // tile_dim0 = 64
            g1[4] = 64;                               // tile_dim1 = 64
            g1[5] = 4096;                             // tensor_dim0_stride = N
            g1[6] = 0;
            g1[7] = 0;
            __builtin_amdgcn_tensor_load_to_lds(g0, g1, g2, g3, g4, 0);
            __builtin_amdgcn_s_wait_tensorcnt(0);
        }
#else
        {
            int c = tid >> 2, i16 = (tid & 3) * 16;
            *(v8bf*)&Ht[c * 64 + i16]     = *(const v8bf*)&H[(size_t)c * N_ + t + i16];
            *(v8bf*)&Ht[c * 64 + i16 + 8] = *(const v8bf*)&H[(size_t)c * N_ + t + i16 + 8];
        }
#endif
        if (tid < 64 && t + 64 < N_)                  // prefetch next K tile
            __builtin_prefetch(&F[(size_t)tid * N_ + t + 64], 0, 1);
        __syncthreads();

        // S[i,j] = sum_c F[c,i] G[c,j]   (64x64 scores, K=64)
        v8f s0 = {0.f,0.f,0.f,0.f,0.f,0.f,0.f,0.f};
        v8f s1 = s0;
#pragma unroll
        for (int k0 = 0; k0 < 64; k0 += 32) {
            v16bf bb = frag_B(GtT, scol * 16, k0, 72);
            s0 = wmma_bf16(frag_A(FtT, srow * 16, k0, 72), bb, s0);
            s1 = wmma_bf16(frag_A(FtT, (srow + 2) * 16, k0, 72), bb, s1);
        }
#pragma unroll
        for (int r = 0; r < 8; ++r) {
            Sm[(scol * 16 + col) * 65 + srow * 16 + r + 8 * hh]       = s0[r];
            Sm[(scol * 16 + col) * 65 + (srow + 2) * 16 + r + 8 * hh] = s1[r];
        }
        __syncthreads();

        // online softmax over keys (axis i), 4 threads per column j, 16 rows each
        float pm = -3.0e38f;
#pragma unroll
        for (int i = 0; i < 16; ++i) pm = fmaxf(pm, Sm[j * 65 + q * 16 + i]);
        red[q][j] = pm;
        __syncthreads();
        if (tid < 64) {
            float mx = fmaxf(fmaxf(red[0][j], red[1][j]),
                             fmaxf(red[2][j], red[3][j]));
            mx = fmaxf(mx, mrow[j]);
            arow[j] = __expf(mrow[j] - mx);
            mrow[j] = mx;
        }
        __syncthreads();
        {
            float mx = mrow[j];
            float ps = 0.f;
#pragma unroll
            for (int i = 0; i < 16; ++i) {
                float p = __expf(Sm[j * 65 + q * 16 + i] - mx);
                PtT[j * 72 + q * 16 + i] = (bf16)p;
                ps += p;
            }
            red[q][j] = ps;
        }
        __syncthreads();

        // O := O*alpha + H P (runs concurrently with the l update)
        if (tid < 64)
            lrow[j] = lrow[j] * arow[j] +
                      (red[0][j] + red[1][j] + red[2][j] + red[3][j]);
        float al = arow[jt * 16 + col];
#pragma unroll
        for (int r = 0; r < 8; ++r) { o0[r] *= al; o1[r] *= al; }
#pragma unroll
        for (int k0 = 0; k0 < 64; k0 += 32) {
            v16bf pb = frag_B(PtT, jt * 16, k0, 72);
            o0 = wmma_bf16(frag_A(Ht, ct * 16, k0, 64), pb, o0);
            o1 = wmma_bf16(frag_A(Ht, (ct + 2) * 16, k0, 64), pb, o1);
        }
    }
    __syncthreads();

    float* OutB = SA8 + (size_t)b * C8_ * N_;
    float linv = 1.f / lrow[jt * 16 + col];
#pragma unroll
    for (int r = 0; r < 8; ++r) {
        int m = r + 8 * hh;
        OutB[(size_t)(ct * 16 + m) * N_ + jb + jt * 16 + col]       = o0[r] * linv;
        OutB[(size_t)((ct + 2) * 16 + m) * N_ + jb + jt * 16 + col] = o1[r] * linv;
    }
}

// ---------------------------------------------------------------------------
// Kernel 3: sa = Wv (512x64) @ sa8 (64x4096); out = gamma*sa + x.
// d_out = [out | sa]. Grid (N/64, C/64, B), 256 threads.
// ---------------------------------------------------------------------------
__global__ __launch_bounds__(256)
void vproj_kernel(const float* __restrict__ x,
                  const float* __restrict__ Wv,
                  const float* __restrict__ gamma,
                  const float* __restrict__ SA8,
                  float* __restrict__ out) {
    const int jb = blockIdx.x * 64;
    const int ob = blockIdx.y * 64;
    const int b  = blockIdx.z;
    __shared__ __align__(16) bf16 At [64 * 64];   // Wv  [o][k] ld 64
    __shared__ __align__(16) bf16 BtT[64 * 72];   // sa8 [j][k] ld 72

    const int tid  = threadIdx.x;
    const int wave = tid >> 5;
    const int lane = tid & 31;
    const int col  = lane & 15, hh = lane >> 4;
    const int ct = wave >> 2, jt = wave & 3;
    const float* SA = SA8 + (size_t)b * C8_ * N_;

#pragma unroll
    for (int e = 0; e < 4; ++e) {                 // A: 64x64 fp32 -> bf16
        int ch = tid + e * 256;
        int r = ch >> 4, c4 = (ch & 15) * 4;
        float4 w4 = *(const float4*)&Wv[(size_t)(ob + r) * C8_ + c4];
        *(v4bf*)&At[r * 64 + c4] = pack4(w4.x, w4.y, w4.z, w4.w);
    }
#pragma unroll
    for (int e = 0; e < 4; ++e) {                 // B: transposed scatter
        int ch = tid + e * 256;
        int k = ch >> 4, j4 = (ch & 15) * 4;
        float4 v = *(const float4*)&SA[(size_t)k * N_ + jb + j4];
        BtT[(j4 + 0) * 72 + k] = (bf16)v.x;
        BtT[(j4 + 1) * 72 + k] = (bf16)v.y;
        BtT[(j4 + 2) * 72 + k] = (bf16)v.z;
        BtT[(j4 + 3) * 72 + k] = (bf16)v.w;
    }
    __syncthreads();

    v8f a0 = {0.f,0.f,0.f,0.f,0.f,0.f,0.f,0.f};
    v8f a1 = a0;
#pragma unroll
    for (int k0 = 0; k0 < 64; k0 += 32) {
        v16bf bfrag = frag_B(BtT, jt * 16, k0, 72);
        a0 = wmma_bf16(frag_A(At, ct * 16, k0, 64), bfrag, a0);
        a1 = wmma_bf16(frag_A(At, (ct + 2) * 16, k0, 64), bfrag, a1);
    }

    const float g = gamma[0];
    float* outp = out;
    float* sap  = out + (size_t)B_ * C_ * N_;
#pragma unroll
    for (int r = 0; r < 8; ++r) {
        int m = r + 8 * hh;
        size_t i0 = ((size_t)b * C_ + (ob + ct * 16 + m)) * N_ + jb + jt * 16 + col;
        size_t i1 = ((size_t)b * C_ + (ob + (ct + 2) * 16 + m)) * N_ + jb + jt * 16 + col;
        float s0 = a0[r], s1 = a1[r];
        sap[i0]  = s0;
        sap[i1]  = s1;
        outp[i0] = fmaf(s0, g, x[i0]);
        outp[i1] = fmaf(s1, g, x[i1]);
    }
}

// ---------------------------------------------------------------------------
extern "C" void kernel_launch(void* const* d_in, const int* in_sizes, int n_in,
                              void* d_out, int out_size, void* d_ws, size_t ws_size,
                              hipStream_t stream) {
    const float* x     = (const float*)d_in[0];
    const float* Wf    = (const float*)d_in[1];
    const float* Wg    = (const float*)d_in[2];
    const float* Wh    = (const float*)d_in[3];
    const float* Wv    = (const float*)d_in[4];
    const float* gamma = (const float*)d_in[5];

    bf16*  FGH = (bf16*)d_ws;                                             // 3*B*64*N bf16
    float* SA8 = (float*)((char*)d_ws + (size_t)3 * B_ * C8_ * N_ * 2);   // B*64*N fp32
    float* out = (float*)d_out;

    proj_kernel <<<dim3(N_ / 64, 3, B_),       256, 0, stream>>>(x, Wf, Wg, Wh, FGH);
    attn_kernel <<<dim3(N_ / 64, B_),          256, 0, stream>>>(FGH, SA8);
    vproj_kernel<<<dim3(N_ / 64, C_ / 64, B_), 256, 0, stream>>>(x, Wv, gamma, SA8, out);
}
